// Block_24498493456699
// MI455X (gfx1250) — matrix operand order repaired and checked by reading, not compiled, not run on hardware
//
#include <hip/hip_runtime.h>
#include <cmath>

// ---------------------------------------------------------------------------
// ViT transformer block on gfx1250 (MI455X), wave32 + WMMA f16 (f32 accum).
// B=16, N=1024, D=768, H=12, HD=64, HID=3072.  All GEMMs are x*W^T with W
// stored [out,in], so A and B WMMA fragments both read contiguous K runs.
// GEMM uses double-buffered LDS tiles + global prefetch of the tile after
// next; attention is flash-style with V^T/P staged in LDS.
// ---------------------------------------------------------------------------

typedef __attribute__((ext_vector_type(16))) _Float16 v16h;
typedef __attribute__((ext_vector_type(8)))  _Float16 h8;
typedef __attribute__((ext_vector_type(2)))  _Float16 h2;
typedef __attribute__((ext_vector_type(8)))  float    v8f;

static constexpr int Bsz  = 16;
static constexpr int Ntok = 1024;
static constexpr int D    = 768;
static constexpr int Hh   = 12;
static constexpr int HD   = 64;
static constexpr int HID  = 3072;
static constexpr int MROW = Bsz * Ntok;          // 16384 token rows
static constexpr int QKVC = 3 * D;               // 2304

__device__ __forceinline__ v8f wmma16(v16h a, v16h b, v8f c) {
  // emits v_wmma_f32_16x16x32_f16
  return __builtin_amdgcn_wmma_f32_16x16x32_f16(false, a, false, b,
                                                (short)0, c, false, false);
}

union FragU { v16h v; h8 c[2]; };

// ---------------------------------------------------------------------------
// Elementwise f32 -> f16 cast (weights)
// ---------------------------------------------------------------------------
__global__ __launch_bounds__(256) void cast_f16_kernel(const float* __restrict__ s,
                                                       _Float16* __restrict__ d, int n) {
  int i = blockIdx.x * 256 + threadIdx.x;
  if (i < n) d[i] = (_Float16)s[i];
}

// ---------------------------------------------------------------------------
// LayerNorm over D=768, output f16.  One block per token row, 3 elems/thread.
// ---------------------------------------------------------------------------
__global__ __launch_bounds__(256) void ln_f16_kernel(const float* __restrict__ x,
                                                     const float* __restrict__ w,
                                                     const float* __restrict__ b,
                                                     _Float16* __restrict__ out) {
  int row = blockIdx.x, tid = threadIdx.x;
  const float* xr = x + (size_t)row * D;
  float v0 = xr[tid], v1 = xr[tid + 256], v2 = xr[tid + 512];
  __shared__ float red[256];
  red[tid] = v0 + v1 + v2;
  __syncthreads();
  for (int st = 128; st > 0; st >>= 1) {
    if (tid < st) red[tid] += red[tid + st];
    __syncthreads();
  }
  float mu = red[0] * (1.0f / D);
  __syncthreads();
  float d0 = v0 - mu, d1 = v1 - mu, d2 = v2 - mu;
  red[tid] = d0 * d0 + d1 * d1 + d2 * d2;
  __syncthreads();
  for (int st = 128; st > 0; st >>= 1) {
    if (tid < st) red[tid] += red[tid + st];
    __syncthreads();
  }
  float rs = rsqrtf(red[0] * (1.0f / D) + 1e-5f);
  _Float16* o = out + (size_t)row * D;
  o[tid]       = (_Float16)(d0 * rs * w[tid]       + b[tid]);
  o[tid + 256] = (_Float16)(d1 * rs * w[tid + 256] + b[tid + 256]);
  o[tid + 512] = (_Float16)(d2 * rs * w[tid + 512] + b[tid + 512]);
}

// ---------------------------------------------------------------------------
// WMMA GEMM: C[M,N] = A[M,K] * W[N,K]^T (+epilogue).  128x128 block tile,
// K-step 32, 8 waves in 4(m) x 2(n); each wave -> 32x64 = 8 WMMA tiles.
// Double-buffered LDS (pitch 40 halves keeps 16B-aligned chunks), one
// barrier per K-step, prefetch of the tile after next.
// EPI: 0 = f16 store, 1 = x + (acc+bias)*gamma -> f32, 2 = GELU -> f16.
// ---------------------------------------------------------------------------
template <int EPI>
__global__ __launch_bounds__(256) void gemm_f16_kernel(
    const _Float16* __restrict__ A, const _Float16* __restrict__ W,
    const float* __restrict__ bias, const float* __restrict__ xres,
    const float* __restrict__ gamma, float* __restrict__ outF,
    _Float16* __restrict__ outH, int N, int K) {
  __shared__ _Float16 As[2][128 * 40];
  __shared__ _Float16 Bs[2][128 * 40];
  const int tid = threadIdx.x;
  const int l = tid & 31, w = tid >> 5;
  const int li = l & 15, hi = l >> 4;
  const int wn = w & 1, wm = w >> 1;          // wm 0..3 (M), wn 0..1 (N)
  const int n0 = blockIdx.x * 128, m0 = blockIdx.y * 128;
  const int khalf = hi * 8;                   // A frag K-octet select
  const int kb16  = hi * 8;                   // B frag K-octet select

  v8f acc[2][4];
#pragma unroll
  for (int im = 0; im < 2; ++im)
#pragma unroll
    for (int in = 0; in < 4; ++in)
#pragma unroll
      for (int e = 0; e < 8; ++e) acc[im][in][e] = 0.0f;

  auto stage = [&](int buf, int kt) {
#pragma unroll
    for (int it = 0; it < 2; ++it) {          // 512 16B chunks / 256 threads
      int i = tid + it * 256;
      int row = i >> 2, c = (i & 3) * 8;
      *(h8*)(&As[buf][row * 40 + c]) =
          *(const h8*)(A + (size_t)(m0 + row) * K + kt + c);
      *(h8*)(&Bs[buf][row * 40 + c]) =
          *(const h8*)(W + (size_t)(n0 + row) * K + kt + c);
    }
  };

  const int nk = K / 32;
  stage(0, 0);
  for (int t = 0; t < nk; ++t) {
    __syncthreads();                          // staged tile t visible
    const int cur = t & 1;
    if (t + 1 < nk) stage(cur ^ 1, (t + 1) * 32);
    if (t + 2 < nk) {                         // prefetch tile t+2 into caches
      int pr = tid & 127;
      __builtin_prefetch(A + (size_t)(m0 + pr) * K + (t + 2) * 32, 0, 1);
      __builtin_prefetch(W + (size_t)(n0 + pr) * K + (t + 2) * 32, 0, 1);
    }
    v16h af[2], bf[4];
#pragma unroll
    for (int im = 0; im < 2; ++im) {
      int row = wm * 32 + im * 16 + li;
      FragU u;
      u.c[0] = *(const h8*)(&As[cur][row * 40 + khalf]);
      u.c[1] = *(const h8*)(&As[cur][row * 40 + khalf + 16]);
      af[im] = u.v;
    }
#pragma unroll
    for (int in = 0; in < 4; ++in) {
      int col = wn * 64 + in * 16 + li;
      FragU u;
      u.c[0] = *(const h8*)(&Bs[cur][col * 40 + kb16]);
      u.c[1] = *(const h8*)(&Bs[cur][col * 40 + kb16 + 16]);
      bf[in] = u.v;
    }
#pragma unroll
    for (int im = 0; im < 2; ++im)
#pragma unroll
      for (int in = 0; in < 4; ++in)
        acc[im][in] = wmma16(af[im], bf[in], acc[im][in]);
  }

  // Epilogue: C layout -> lane half gives rows v / v+8, lane&15 gives column.
#pragma unroll
  for (int im = 0; im < 2; ++im)
#pragma unroll
    for (int in = 0; in < 4; ++in)
#pragma unroll
      for (int v = 0; v < 8; ++v) {
        int m = m0 + wm * 32 + im * 16 + v + 8 * hi;
        int n = n0 + wn * 64 + in * 16 + li;
        float val = acc[im][in][v];
        if (EPI == 0) {
          outH[(size_t)m * N + n] = (_Float16)val;
        } else if (EPI == 1) {
          outF[(size_t)m * N + n] =
              xres[(size_t)m * N + n] + (val + bias[n]) * gamma[n];
        } else {
          float g = val + bias[n];
          g = 0.5f * g * (1.0f + erff(g * 0.70710678118f));
          outH[(size_t)m * N + n] = (_Float16)g;
        }
      }
}

// ---------------------------------------------------------------------------
// Flash attention: one block per (b, h, 128 q-rows).  Q A-frags in registers
// (loaded straight from qkv), K B-frags straight from global (contiguous HD),
// V^T and P staged in LDS.  Online softmax with half-wave shfl reductions.
// ---------------------------------------------------------------------------
__global__ __launch_bounds__(256) void attn_kernel(const _Float16* __restrict__ qkv,
                                                   _Float16* __restrict__ o) {
  const int tid = threadIdx.x;
  const int l = tid & 31, w = tid >> 5;
  const int li = l & 15, hi = l >> 4;
  const int blk = blockIdx.x;
  const int qb = blk & 7, bh = blk >> 3;
  const int h = bh % Hh, b = bh / Hh;
  const int q0 = qb * 128;
  const size_t pitch = QKVC;
  const _Float16* qbase = qkv + (size_t)(b * Ntok) * pitch + h * HD;
  const _Float16* kbase = qbase + D;
  const _Float16* vbase = qbase + 2 * D;

  __shared__ _Float16 VT[64 * 136];    // V^T: [d][key], pitch 136
  __shared__ _Float16 Ps[128 * 136];   // P:   [row][key], pitch 136

  // Preload Q A-fragments for this wave's 16 rows (K = HD = 64 -> 2 k-steps)
  v16h aq[2];
  {
    int row = q0 + w * 16 + li;
    const _Float16* p = qbase + (size_t)row * pitch + hi * 8;
#pragma unroll
    for (int ks = 0; ks < 2; ++ks) {
      FragU u;
      u.c[0] = *(const h8*)(p + ks * 32);
      u.c[1] = *(const h8*)(p + ks * 32 + 16);
      aq[ks] = u.v;
    }
  }

  v8f oacc[4];
#pragma unroll
  for (int t = 0; t < 4; ++t)
#pragma unroll
    for (int e = 0; e < 8; ++e) oacc[t][e] = 0.0f;
  float mrow[8], lrow[8];
#pragma unroll
  for (int v = 0; v < 8; ++v) { mrow[v] = -3.0e38f; lrow[v] = 0.0f; }

  for (int j = 0; j < 8; ++j) {
    const int k0 = j * 128;
    __syncthreads();                            // protect VT/Ps from prev iter
    // Stage V^T: each thread handles a pair of adjacent keys for one d-chunk
    // so transposed writes pack {V[k][d],V[k+1][d]} into one b32 store.
#pragma unroll
    for (int it = 0; it < 2; ++it) {
      int idx = tid + it * 256;                 // 512 pair-chunks
      int kp = idx >> 3, dc = (idx & 7) * 8;    // kp 0..63, dc 0..56
      int key = kp * 2;
      h8 va = *(const h8*)(vbase + (size_t)(k0 + key)     * pitch + dc);
      h8 vb = *(const h8*)(vbase + (size_t)(k0 + key + 1) * pitch + dc);
#pragma unroll
      for (int e = 0; e < 8; ++e) {
        h2 pk = {va[e], vb[e]};
        *(h2*)(&VT[(dc + e) * 136 + key]) = pk;
      }
    }

    // S = Q * K^T for this wave's 16 rows x 128 keys
    v8f sacc[8];
#pragma unroll
    for (int t = 0; t < 8; ++t)
#pragma unroll
      for (int e = 0; e < 8; ++e) sacc[t][e] = 0.0f;
#pragma unroll
    for (int ks = 0; ks < 2; ++ks)
#pragma unroll
      for (int t = 0; t < 8; ++t) {
        int key = k0 + t * 16 + li;
        const _Float16* p = kbase + (size_t)key * pitch + ks * 32 + hi * 8;
        FragU u;
        u.c[0] = *(const h8*)p;
        u.c[1] = *(const h8*)(p + 16);
        sacc[t] = wmma16(aq[ks], u.v, sacc[t]);
      }

    // Online softmax (rows v / v+8 live in lane halves; reduce across 16 lanes)
#pragma unroll
    for (int v = 0; v < 8; ++v) {
      float mx = -3.0e38f;
#pragma unroll
      for (int t = 0; t < 8; ++t) mx = fmaxf(mx, sacc[t][v] * 0.125f);
      for (int sh = 8; sh >= 1; sh >>= 1) mx = fmaxf(mx, __shfl_xor(mx, sh));
      float mnew = fmaxf(mrow[v], mx);
      float corr = __expf(mrow[v] - mnew);
      float sum = 0.0f;
      int row = w * 16 + v + 8 * hi;
#pragma unroll
      for (int t = 0; t < 8; ++t) {
        float p = __expf(sacc[t][v] * 0.125f - mnew);
        sum += p;
        Ps[row * 136 + t * 16 + li] = (_Float16)p;
      }
      for (int sh = 8; sh >= 1; sh >>= 1) sum += __shfl_xor(sum, sh);
      lrow[v] = lrow[v] * corr + sum;
      mrow[v] = mnew;
#pragma unroll
      for (int t = 0; t < 4; ++t) oacc[t][v] *= corr;
    }
    __syncthreads();                            // Ps + VT visible

    // O += P * V   (contraction over 128 keys -> 4 k-steps)
#pragma unroll
    for (int ks = 0; ks < 4; ++ks) {
      FragU ua;
      int row = w * 16 + li;
      const _Float16* pp = &Ps[row * 136 + ks * 32 + hi * 8];
      ua.c[0] = *(const h8*)pp;
      ua.c[1] = *(const h8*)(pp + 16);
#pragma unroll
      for (int t = 0; t < 4; ++t) {
        FragU ub;
        const _Float16* vp = &VT[(t * 16 + li) * 136 + ks * 32 + hi * 8];
        ub.c[0] = *(const h8*)vp;
        ub.c[1] = *(const h8*)(vp + 16);
        oacc[t] = wmma16(ua.v, ub.v, oacc[t]);
      }
    }
  }

  // Normalize and store to o[b*N+row][h*64+d] (f16)
  _Float16* obase = o + (size_t)(b * Ntok + q0) * D + h * HD;
#pragma unroll
  for (int t = 0; t < 4; ++t)
#pragma unroll
    for (int v = 0; v < 8; ++v) {
      int row = w * 16 + v + 8 * hi;
      float val = oacc[t][v] / lrow[v];
      obase[(size_t)row * D + t * 16 + li] = (_Float16)val;
    }
}

// ---------------------------------------------------------------------------
// Host-side orchestration
// ---------------------------------------------------------------------------
extern "C" void kernel_launch(void* const* d_in, const int* in_sizes, int n_in,
                              void* d_out, int out_size, void* d_ws, size_t ws_size,
                              hipStream_t stream) {
  const float* x      = (const float*)d_in[0];
  const float* ln1_w  = (const float*)d_in[1];
  const float* ln1_b  = (const float*)d_in[2];
  const float* qkv_w  = (const float*)d_in[3];
  const float* proj_w = (const float*)d_in[4];
  const float* proj_b = (const float*)d_in[5];
  const float* gamma1 = (const float*)d_in[6];
  const float* ln2_w  = (const float*)d_in[7];
  const float* ln2_b  = (const float*)d_in[8];
  const float* fc1_w  = (const float*)d_in[9];
  const float* fc1_b  = (const float*)d_in[10];
  const float* fc2_w  = (const float*)d_in[11];
  const float* fc2_b  = (const float*)d_in[12];
  const float* gamma2 = (const float*)d_in[13];
  float* out = (float*)d_out;

  char* ws = (char*)d_ws;
  size_t off = 0;
  auto alloc = [&](size_t bytes) {
    size_t r = off;
    off += (bytes + 255) & ~(size_t)255;
    return r;
  };
  _Float16* w_qkv_h  = (_Float16*)(ws + alloc((size_t)QKVC * D * 2));
  _Float16* w_proj_h = (_Float16*)(ws + alloc((size_t)D * D * 2));
  _Float16* w_fc1_h  = (_Float16*)(ws + alloc((size_t)HID * D * 2));
  _Float16* w_fc2_h  = (_Float16*)(ws + alloc((size_t)D * HID * 2));
  _Float16* h_h      = (_Float16*)(ws + alloc((size_t)MROW * D * 2));      // LN out (reused)
  _Float16* qkv_h    = (_Float16*)(ws + alloc((size_t)MROW * QKVC * 2));
  _Float16* o_h      = (_Float16*)(ws + alloc((size_t)MROW * D * 2));      // attention out
  float*    x1       = (float*)(ws + alloc((size_t)MROW * D * 4));         // post-attn residual
  _Float16* g_h      = (_Float16*)(ws + alloc((size_t)MROW * HID * 2));    // GELU out

  // 1) weight casts
  auto cast = [&](const float* s, _Float16* d, int n) {
    cast_f16_kernel<<<(n + 255) / 256, 256, 0, stream>>>(s, d, n);
  };
  cast(qkv_w,  w_qkv_h,  QKVC * D);
  cast(proj_w, w_proj_h, D * D);
  cast(fc1_w,  w_fc1_h,  HID * D);
  cast(fc2_w,  w_fc2_h,  D * HID);

  // 2) LN1
  ln_f16_kernel<<<MROW, 256, 0, stream>>>(x, ln1_w, ln1_b, h_h);

  // 3) QKV GEMM: [16384,2304] = h * qkv_w^T
  gemm_f16_kernel<0><<<dim3(QKVC / 128, MROW / 128), 256, 0, stream>>>(
      h_h, w_qkv_h, nullptr, nullptr, nullptr, nullptr, qkv_h, QKVC, D);

  // 4) attention (flash over 8 key blocks), grid = B*H*(N/128)
  attn_kernel<<<Bsz * Hh * (Ntok / 128), 256, 0, stream>>>(qkv_h, o_h);

  // 5) proj GEMM + residual: x1 = x + (o*proj_w^T + proj_b)*gamma1
  gemm_f16_kernel<1><<<dim3(D / 128, MROW / 128), 256, 0, stream>>>(
      o_h, w_proj_h, proj_b, x, gamma1, x1, nullptr, D, D);

  // 6) LN2
  ln_f16_kernel<<<MROW, 256, 0, stream>>>(x1, ln2_w, ln2_b, h_h);

  // 7) FC1 + exact GELU
  gemm_f16_kernel<2><<<dim3(HID / 128, MROW / 128), 256, 0, stream>>>(
      h_h, w_fc1_h, fc1_b, nullptr, nullptr, nullptr, g_h, HID, D);

  // 8) FC2 + residual: out = x1 + (g*fc2_w^T + fc2_b)*gamma2
  gemm_f16_kernel<1><<<dim3(D / 128, MROW / 128), 256, 0, stream>>>(
      g_h, w_fc2_h, fc2_b, x1, gamma2, out, nullptr, D, HID);

  (void)in_sizes; (void)n_in; (void)out_size; (void)ws_size;
}
